// MEGNet_28329604284558
// MI455X (gfx1250) — matrix operand order repaired and checked by reading, not compiled, not run on hardware
//
#include <hip/hip_runtime.h>
#include <hip/hip_bf16.h>
#include <math.h>

#define HDIM 64
#define NGATE 3

typedef __attribute__((ext_vector_type(16))) _Float16 v16h;
typedef __attribute__((ext_vector_type(8)))  float    v8f;

// ---------------------------------------------------------------------------
// WMMA fragment helpers (v_wmma_f32_16x16x32_f16, wave32)
//
// A (16x32 f16): lane l (l<16): row M=l, k in {kb..kb+7, kb+16..kb+23}, kb=0
//                lane l (>=16): row M=l-16, kb=8  (per CDNA5 ISA 7.12.2)
// B (32x16 f16): lane l: col N=l&15; same k split as A:
//                k in {kb..kb+7, kb+16..kb+23}, kb = kw + 8*(l>>4).
// C (16x16 f32): c[j] = C[M = j + 8*(l>>4)][N = l&15]
// ---------------------------------------------------------------------------

__device__ __forceinline__ v16h a_frag_from_f32(const float* __restrict__ Xrow,
                                                int kw, int lane) {
  const int kb = kw + ((lane >> 4) << 3);
  const float4* p = (const float4*)(Xrow + kb);
  float4 u0 = p[0];
  float4 u1 = p[1];
  const float4* q = (const float4*)(Xrow + kb + 16);
  float4 u2 = q[0];
  float4 u3 = q[1];
  v16h a;
  a[0] = (_Float16)u0.x;  a[1] = (_Float16)u0.y;
  a[2] = (_Float16)u0.z;  a[3] = (_Float16)u0.w;
  a[4] = (_Float16)u1.x;  a[5] = (_Float16)u1.y;
  a[6] = (_Float16)u1.z;  a[7] = (_Float16)u1.w;
  a[8] = (_Float16)u2.x;  a[9] = (_Float16)u2.y;
  a[10] = (_Float16)u2.z; a[11] = (_Float16)u2.w;
  a[12] = (_Float16)u3.x; a[13] = (_Float16)u3.y;
  a[14] = (_Float16)u3.z; a[15] = (_Float16)u3.w;
  return a;
}

// B fragment from LDS, weights stored output-col-major: Wl[col*64 + k]
__device__ __forceinline__ v16h b_frag_from_lds(const _Float16* __restrict__ Wl,
                                                int col, int kw, int lane) {
  typedef __attribute__((ext_vector_type(8))) _Float16 v8h_t;
  const int kb = kw + ((lane >> 4) << 3);
  v8h_t lo = *(const v8h_t*)(Wl + col * HDIM + kb);
  v8h_t hi = *(const v8h_t*)(Wl + col * HDIM + kb + 16);
  return __builtin_shufflevector(lo, hi, 0, 1, 2, 3, 4, 5, 6, 7, 8, 9, 10, 11, 12, 13, 14, 15);
}

__device__ __forceinline__ v8f wmma_f16(v16h a, v16h b, v8f c) {
  return __builtin_amdgcn_wmma_f32_16x16x32_f16(false, a, false, b,
                                                (short)0, c, false, false);
}

__device__ __forceinline__ float sigmoidf_fast(float x) {
  return 1.0f / (1.0f + __expf(-x));
}

// ---------------------------------------------------------------------------
// Kernels
// ---------------------------------------------------------------------------

__global__ void embed_kernel(const int* __restrict__ x,
                             const float* __restrict__ emb,
                             float* __restrict__ h, int total) {
  int t = blockIdx.x * blockDim.x + threadIdx.x;
  if (t >= total) return;
  int n = t >> 6, c = t & 63;
  h[t] = emb[x[n] * HDIM + c];
}

__global__ void zero_kernel(float* __restrict__ p, size_t n) {
  size_t t = (size_t)blockIdx.x * blockDim.x + threadIdx.x;
  if (t < n) p[t] = 0.0f;
}

// Y[N x 64] = X[N x 64] @ W[64 x 64]   (W row-major k x j)
__global__ void __launch_bounds__(256) gemm_node_kernel(
    const float* __restrict__ X, const float* __restrict__ W,
    float* __restrict__ Y, int nRowTiles) {
  __shared__ _Float16 WsT[HDIM * HDIM];   // WsT[j*64 + k] = W[k*64 + j], f16
  for (int i = threadIdx.x; i < HDIM * HDIM; i += blockDim.x) {
    int k = i >> 6, j = i & 63;
    WsT[j * HDIM + k] = (_Float16)W[i];
  }
  __syncthreads();

  const int wave = blockIdx.x * (blockDim.x >> 5) + (threadIdx.x >> 5);
  if (wave >= nRowTiles) return;                 // wave-uniform: EXEC stays full
  const int lane = threadIdx.x & 31;
  const int row0 = wave * 16;

  const float* Xrow = X + (size_t)(row0 + (lane & 15)) * HDIM;
  const v16h a0 = a_frag_from_f32(Xrow, 0, lane);
  const v16h a1 = a_frag_from_f32(Xrow, 32, lane);

  const int colL = lane & 15;
  const int rbase = row0 + ((lane >> 4) << 3);
#pragma unroll
  for (int ct = 0; ct < 4; ++ct) {
    const int col = ct * 16 + colL;
    v16h b0 = b_frag_from_lds(WsT, col, 0, lane);
    v16h b1 = b_frag_from_lds(WsT, col, 32, lane);
    v8f c = {};
    c = wmma_f16(a0, b0, c);
    c = wmma_f16(a1, b1, c);
#pragma unroll
    for (int j = 0; j < 8; ++j)
      Y[(size_t)(rbase + j) * HDIM + col] = c[j];
  }
}

// agg[dst[e]] += m[src[e]]   (64 floats per edge, 4 floats per thread)
__global__ void scatter_kernel(const float* __restrict__ m,
                               const int* __restrict__ src,
                               const int* __restrict__ dst,
                               float* __restrict__ agg, int E) {
  long long t = (long long)blockIdx.x * blockDim.x + threadIdx.x;
  int e = (int)(t >> 4);
  if (e >= E) return;
  int c4 = (int)(t & 15) << 2;
  int s = src[e], d = dst[e];
  float4 v = *(const float4*)(m + (size_t)s * HDIM + c4);
  float* o = agg + (size_t)d * HDIM + c4;
  atomicAdd(o + 0, v.x);
  atomicAdd(o + 1, v.y);
  atomicAdd(o + 2, v.z);
  atomicAdd(o + 3, v.w);
}

// Fused GRU + ReLU. Each wave: 16 rows x 16 cols, computes all 6 gate tiles.
__global__ void __launch_bounds__(256) gru_fused_kernel(
    const float* __restrict__ agg, const float* __restrict__ h,
    const float* __restrict__ wih, const float* __restrict__ whh,  // 192x64
    const float* __restrict__ bih, const float* __restrict__ bhh,  // 192
    float* __restrict__ hout, int nTasks) {
  __shared__ _Float16 WihS[NGATE * HDIM * HDIM];   // 12288 halfs
  __shared__ _Float16 WhhS[NGATE * HDIM * HDIM];
  for (int i = threadIdx.x; i < NGATE * HDIM * HDIM; i += blockDim.x) {
    WihS[i] = (_Float16)wih[i];   // already output-col-major: wih[j][k]
    WhhS[i] = (_Float16)whh[i];
  }
  __syncthreads();

  const int task = blockIdx.x * (blockDim.x >> 5) + (threadIdx.x >> 5);
  if (task >= nTasks) return;                    // wave-uniform
  const int lane = threadIdx.x & 31;
  const int row0 = (task >> 2) * 16;
  const int c0 = (task & 3) * 16;

  const float* Arow = agg + (size_t)(row0 + (lane & 15)) * HDIM;
  const float* Hrow = h + (size_t)(row0 + (lane & 15)) * HDIM;
  const v16h aA0 = a_frag_from_f32(Arow, 0, lane);
  const v16h aA1 = a_frag_from_f32(Arow, 32, lane);
  const v16h aH0 = a_frag_from_f32(Hrow, 0, lane);
  const v16h aH1 = a_frag_from_f32(Hrow, 32, lane);

  const int col = c0 + (lane & 15);
  v8f gi[NGATE], gh[NGATE];
#pragma unroll
  for (int g = 0; g < NGATE; ++g) {
    const int wrow = g * HDIM + col;   // output column index within 192
    v16h bi0 = b_frag_from_lds(WihS, wrow, 0, lane);
    v16h bi1 = b_frag_from_lds(WihS, wrow, 32, lane);
    v8f ci = {};
    ci = wmma_f16(aA0, bi0, ci);
    ci = wmma_f16(aA1, bi1, ci);
    v16h bh0 = b_frag_from_lds(WhhS, wrow, 0, lane);
    v16h bh1 = b_frag_from_lds(WhhS, wrow, 32, lane);
    v8f ch = {};
    ch = wmma_f16(aH0, bh0, ch);
    ch = wmma_f16(aH1, bh1, ch);
    const float bi = bih[wrow], bh = bhh[wrow];
#pragma unroll
    for (int j = 0; j < 8; ++j) { ci[j] += bi; ch[j] += bh; }
    gi[g] = ci;
    gh[g] = ch;
  }

  const int rbase = row0 + ((lane >> 4) << 3);
#pragma unroll
  for (int j = 0; j < 8; ++j) {
    float hold = h[(size_t)(rbase + j) * HDIM + col];
    float r = sigmoidf_fast(gi[0][j] + gh[0][j]);
    float z = sigmoidf_fast(gi[1][j] + gh[1][j]);
    float n = tanhf(gi[2][j] + r * gh[2][j]);
    float hn = (1.0f - z) * n + z * hold;
    hout[(size_t)(rbase + j) * HDIM + col] = fmaxf(hn, 0.0f);
  }
}

__global__ void pool_kernel(const float* __restrict__ h,
                            const int* __restrict__ batch,
                            float* __restrict__ sums,
                            float* __restrict__ counts, int N) {
  long long t = (long long)blockIdx.x * blockDim.x + threadIdx.x;
  int n = (int)(t >> 4);
  if (n >= N) return;
  int c4 = (int)(t & 15) << 2;
  int b = batch[n];
  float4 v = *(const float4*)(h + (size_t)n * HDIM + c4);
  float* o = sums + (size_t)b * HDIM + c4;
  atomicAdd(o + 0, v.x);
  atomicAdd(o + 1, v.y);
  atomicAdd(o + 2, v.z);
  atomicAdd(o + 3, v.w);
  if (c4 == 0) atomicAdd(counts + b, 1.0f);
}

__global__ void __launch_bounds__(256) mlp_kernel(
    const float* __restrict__ sums, const float* __restrict__ counts,
    const float* __restrict__ fc1W, const float* __restrict__ fc1b,
    const float* __restrict__ fc2W, const float* __restrict__ fc2b,
    const float* __restrict__ fc3W, const float* __restrict__ fc3b,
    float* __restrict__ out, int B) {
  int b = blockIdx.x * blockDim.x + threadIdx.x;
  if (b >= B) return;
  float inv = 1.0f / fmaxf(counts[b], 1.0f);
  float p[64];
#pragma unroll
  for (int c = 0; c < 64; ++c) p[c] = sums[(size_t)b * 64 + c] * inv;
  float y1[32];
  for (int j = 0; j < 32; ++j) {
    float s = fc1b[j];
    for (int k = 0; k < 64; ++k) s += p[k] * fc1W[j * 64 + k];
    y1[j] = fmaxf(s, 0.0f);
  }
  float y2[16];
  for (int j = 0; j < 16; ++j) {
    float s = fc2b[j];
    for (int k = 0; k < 32; ++k) s += y1[k] * fc2W[j * 32 + k];
    y2[j] = fmaxf(s, 0.0f);
  }
  float s = fc3b[0];
  for (int k = 0; k < 16; ++k) s += y2[k] * fc3W[k];
  out[b] = s;
}

// ---------------------------------------------------------------------------

extern "C" void kernel_launch(void* const* d_in, const int* in_sizes, int n_in,
                              void* d_out, int out_size, void* d_ws, size_t ws_size,
                              hipStream_t stream) {
  const int N = 100000, E = 1000000, B = 256, L = 3;

  const int* x = (const int*)d_in[0];
  const int* edge_index = (const int*)d_in[1];        // (2, E)
  const int* batch = (const int*)d_in[3];
  const float* node_emb = (const float*)d_in[4];
  const float* ggc_W = (const float*)d_in[7];         // (L, 64, 64)
  const float* gru_wih = (const float*)d_in[8];       // (L, 192, 64)
  const float* gru_whh = (const float*)d_in[9];
  const float* gru_bih = (const float*)d_in[10];      // (L, 192)
  const float* gru_bhh = (const float*)d_in[11];
  const float* fc1W = (const float*)d_in[12];
  const float* fc1b = (const float*)d_in[13];
  const float* fc2W = (const float*)d_in[14];
  const float* fc2b = (const float*)d_in[15];
  const float* fc3W = (const float*)d_in[16];
  const float* fc3b = (const float*)d_in[17];
  float* out = (float*)d_out;

  const size_t NH = (size_t)N * HDIM;
  float* ws = (float*)d_ws;
  float* h = ws;
  float* h2 = ws + NH;
  float* m = ws + 2 * NH;
  float* agg = ws + 3 * NH;
  float* sums = ws + 4 * NH;          // B*64
  float* counts = sums + (size_t)B * HDIM;

  const int nRowTiles = N / 16;       // 6250 (N is a multiple of 16)

  embed_kernel<<<(N * HDIM + 255) / 256, 256, 0, stream>>>(x, node_emb, h, N * HDIM);

  float* hcur = h;
  float* hnext = h2;
  for (int l = 0; l < L; ++l) {
    gemm_node_kernel<<<(nRowTiles + 7) / 8, 256, 0, stream>>>(
        hcur, ggc_W + (size_t)l * HDIM * HDIM, m, nRowTiles);

    zero_kernel<<<(int)((NH + 255) / 256), 256, 0, stream>>>(agg, NH);

    scatter_kernel<<<(int)(((long long)E * 16 + 255) / 256), 256, 0, stream>>>(
        m, edge_index, edge_index + E, agg, E);

    const int nTasks = nRowTiles * 4;
    gru_fused_kernel<<<(nTasks + 7) / 8, 256, 0, stream>>>(
        agg, hcur,
        gru_wih + (size_t)l * NGATE * HDIM * HDIM,
        gru_whh + (size_t)l * NGATE * HDIM * HDIM,
        gru_bih + (size_t)l * NGATE * HDIM,
        gru_bhh + (size_t)l * NGATE * HDIM,
        hnext, nTasks);

    float* tmp = hcur; hcur = hnext; hnext = tmp;
  }

  zero_kernel<<<((B * HDIM + B) + 255) / 256, 256, 0, stream>>>(sums, (size_t)B * HDIM + B);
  pool_kernel<<<(int)(((long long)N * 16 + 255) / 256), 256, 0, stream>>>(
      hcur, batch, sums, counts, N);
  mlp_kernel<<<1, 256, 0, stream>>>(sums, counts, fc1W, fc1b, fc2W, fc2b,
                                    fc3W, fc3b, out, B);
}
